// ExportWeightBackedLinear_50835232916360
// MI455X (gfx1250) — matrix operand order repaired and checked by reading, not compiled, not run on hardware
//
#include <hip/hip_runtime.h>

// ---- vector types ----
typedef _Float16 v16h __attribute__((ext_vector_type(16)));
typedef _Float16 h8   __attribute__((ext_vector_type(8)));
typedef float    v8f  __attribute__((ext_vector_type(8)));

#define BM   128
#define BN   128
#define BK   32
#define LDA  48            // halves per LDS row: 32 data + 16 pad -> 96B rows (16B aligned, bank-skewed)
#define KDIM 4096
#define NDIM 4096
#define NK   (KDIM / BK)   // 128 K-steps

// ================= Prepass 1: x fp32 -> f16 ==================
// Memory-bound: 67MB read + 33.5MB write. Each thread converts 8 elements.
__global__ __launch_bounds__(256)
void cvt_x_f16_kernel(const float* __restrict__ x, _Float16* __restrict__ xh)
{
    const int g = blockIdx.x * 256 + threadIdx.x;      // h8-group index
    const float4* p = reinterpret_cast<const float4*>(x) + (size_t)g * 2;
    const float4 a = p[0];
    const float4 b = p[1];
    h8 o;
    o[0] = (_Float16)a.x; o[1] = (_Float16)a.y;
    o[2] = (_Float16)a.z; o[3] = (_Float16)a.w;
    o[4] = (_Float16)b.x; o[5] = (_Float16)b.y;
    o[6] = (_Float16)b.z; o[7] = (_Float16)b.w;
    reinterpret_cast<h8*>(xh)[g] = o;
}

// ========= Prepass 2: wh = f16(q - zp)  (exact: |v| <= 255) =========
__global__ __launch_bounds__(256)
void cvt_w_f16_kernel(const int* __restrict__ wq, const int* __restrict__ zp,
                      _Float16* __restrict__ wh)
{
    const int g   = blockIdx.x * 256 + threadIdx.x;    // h8-group index
    const int row = g >> 9;                            // K/8 = 512 groups per row
    const int z   = zp[row];
    const int4* p = reinterpret_cast<const int4*>(wq) + (size_t)g * 2;
    const int4 a = p[0];
    const int4 b = p[1];
    h8 o;
    o[0] = (_Float16)(float)(a.x - z); o[1] = (_Float16)(float)(a.y - z);
    o[2] = (_Float16)(float)(a.z - z); o[3] = (_Float16)(float)(a.w - z);
    o[4] = (_Float16)(float)(b.x - z); o[5] = (_Float16)(float)(b.y - z);
    o[6] = (_Float16)(float)(b.z - z); o[7] = (_Float16)(float)(b.w - z);
    reinterpret_cast<h8*>(wh)[g] = o;
}

// ================= Main GEMM: out = xh @ wh^T * (scale*gamma) + bias*gamma =================
__global__ __launch_bounds__(256)
void ExportWeightBackedLinear_50835232916360_kernel(
        const _Float16* __restrict__ xh,     // [M, K] f16
        const _Float16* __restrict__ wh,     // [N, K] f16 (q - zp)
        const float*    __restrict__ scale,  // [N]
        const float*    __restrict__ bias,   // [N]
        const float*    __restrict__ gamma,  // [N]
        float*          __restrict__ out)    // [M, N] fp32
{
    __shared__ __align__(32) _Float16 sA[2][BM][LDA];
    __shared__ __align__(32) _Float16 sB[2][BN][LDA];

    const int tid  = threadIdx.x;
    const int lane = tid & 31;
    const int wave = tid >> 5;               // 0..7 (wave32)
    const int m0   = blockIdx.x * BM;
    const int n0   = blockIdx.y * BN;

    // 2 waves along M (64 rows each), 4 waves along N (32 cols each)
    const int mW = (wave & 1) * 64;
    const int nW = (wave >> 1) * 32;

    // ---- loader assignment: 2 h8-groups (16B) per thread per matrix ----
    int lrow[2], lc[2];
#pragma unroll
    for (int j = 0; j < 2; ++j) {
        const int g = tid + 256 * j;         // 0..511 groups (128 rows x 4 h8-groups)
        lrow[j] = g >> 2;                    // 0..127
        lc[j]   = (g & 3) * 8;               // k offset (halves)
    }

    // ---- async tile fill: memory -> LDS directly (ASYNCcnt), no VGPR staging ----
    auto issueAsync = [&](int kt, int bufDst) {
        const int kbase = kt * BK;
#pragma unroll
        for (int j = 0; j < 2; ++j) {
            const _Float16* ga = xh + (size_t)(m0 + lrow[j]) * KDIM + kbase + lc[j];
            const unsigned  la = (unsigned)(uintptr_t)&sA[bufDst][lrow[j]][lc[j]];
            asm volatile("global_load_async_to_lds_b128 %0, %1, off"
                         :: "v"(la), "v"((unsigned long long)(uintptr_t)ga)
                         : "memory");
            const _Float16* gb = wh + (size_t)(n0 + lrow[j]) * KDIM + kbase + lc[j];
            const unsigned  lb = (unsigned)(uintptr_t)&sB[bufDst][lrow[j]][lc[j]];
            asm volatile("global_load_async_to_lds_b128 %0, %1, off"
                         :: "v"(lb), "v"((unsigned long long)(uintptr_t)gb)
                         : "memory");
        }
    };

    const v8f vzero = {0.f, 0.f, 0.f, 0.f, 0.f, 0.f, 0.f, 0.f};
    v8f acc[4][2];
#pragma unroll
    for (int i = 0; i < 4; ++i)
#pragma unroll
        for (int j = 0; j < 2; ++j) acc[i][j] = vzero;

    // prologue: fill buffer 0
    issueAsync(0, 0);
    asm volatile("s_wait_asynccnt 0x0" ::: "memory");
    __syncthreads();

    for (int kt = 0; kt < NK; ++kt) {
        const int  buf     = kt & 1;
        const bool hasNext = (kt + 1) < NK;
        if (hasNext) issueAsync(kt + 1, buf ^ 1);   // async fill of other buffer
        if (kt + 2 < NK) {                          // gfx1250 global_prefetch_b8 for K+2 tiles
            __builtin_prefetch(xh + (size_t)(m0 + lrow[0]) * KDIM + (kt + 2) * BK + lc[0], 0, 1);
            __builtin_prefetch(wh + (size_t)(n0 + lrow[0]) * KDIM + (kt + 2) * BK + lc[0], 0, 1);
        }

        // ---- A fragments: ISA 16-bit A layout (lane>>4 selects K phase {0..7,16..23}/{8..15,24..31})
        v16h afrag[4];
        const int klo = (lane >> 4) * 8;     // halves
#pragma unroll
        for (int fm = 0; fm < 4; ++fm) {
            const int row = mW + fm * 16 + (lane & 15);
            h8 lo = *reinterpret_cast<const h8*>(&sA[buf][row][klo]);        // ds_load_b128
            h8 hi = *reinterpret_cast<const h8*>(&sA[buf][row][klo + 16]);   // ds_load_b128
            afrag[fm] = __builtin_shufflevector(
                lo, hi, 0, 1, 2, 3, 4, 5, 6, 7, 8, 9, 10, 11, 12, 13, 14, 15);
        }

        // ---- B fragments: per-lane column of B (KxN), same K phase as A ({0..7,16..23}/{8..15,24..31})
        v16h bfrag[2];
        const int kb = (lane >> 4) * 8;      // halves
#pragma unroll
        for (int fn = 0; fn < 2; ++fn) {
            const int col = nW + fn * 16 + (lane & 15);
            h8 blo = *reinterpret_cast<const h8*>(&sB[buf][col][kb]);        // ds_load_b128
            h8 bhi = *reinterpret_cast<const h8*>(&sB[buf][col][kb + 16]);   // ds_load_b128
            bfrag[fn] = __builtin_shufflevector(
                blo, bhi, 0, 1, 2, 3, 4, 5, 6, 7, 8, 9, 10, 11, 12, 13, 14, 15);
        }

        // ---- 8 WMMAs per wave per K-step ----
#pragma unroll
        for (int fm = 0; fm < 4; ++fm)
#pragma unroll
            for (int fn = 0; fn < 2; ++fn)
                acc[fm][fn] = __builtin_amdgcn_wmma_f32_16x16x32_f16(
                    false, afrag[fm], false, bfrag[fn],
                    (short)0, acc[fm][fn], false, false);

        // async fills must land before anyone reads buf^1 next iteration;
        // __syncthreads only waits DScnt, so wait ASYNCcnt explicitly first.
        if (hasNext) asm volatile("s_wait_asynccnt 0x0" ::: "memory");
        __syncthreads();
    }

    // ---- epilogue: y = acc * (scale*gamma) + bias*gamma ----
#pragma unroll
    for (int fn = 0; fn < 2; ++fn) {
        const int   n  = n0 + nW + fn * 16 + (lane & 15);
        const float g  = gamma[n];
        const float sg = scale[n] * g;
        const float bb = bias[n] * g;
#pragma unroll
        for (int fm = 0; fm < 4; ++fm) {
            const int mBase = m0 + mW + fm * 16 + ((lane >> 4) << 3);
#pragma unroll
            for (int r = 0; r < 8; ++r) {
                out[(size_t)(mBase + r) * NDIM + n] = acc[fm][fn][r] * sg + bb;
            }
        }
    }
}

extern "C" void kernel_launch(void* const* d_in, const int* in_sizes, int n_in,
                              void* d_out, int out_size, void* d_ws, size_t ws_size,
                              hipStream_t stream) {
    const float* x     = (const float*)d_in[0];   // [B,S,DIN] fp32
    const int*   wq    = (const int*)  d_in[1];   // [DOUT,DIN] int32
    const float* scale = (const float*)d_in[2];   // [DOUT]
    const int*   zp    = (const int*)  d_in[3];   // [DOUT]
    const float* bias  = (const float*)d_in[4];   // [DOUT]
    const float* gamma = (const float*)d_in[5];   // [DOUT]
    float*       out   = (float*)d_out;           // [B,S,DOUT] fp32

    const int Mtot = in_sizes[0] / KDIM;          // B*S = 4096

    // workspace layout: xh [M,K] f16 | wh [N,K] f16   (33.5 MB each)
    _Float16* xh = (_Float16*)d_ws;
    _Float16* wh = xh + (size_t)Mtot * KDIM;

    // prepass conversions (memory-bound, ~9us total at 23.3 TB/s)
    {
        const int groups = Mtot * KDIM / 8;
        cvt_x_f16_kernel<<<groups / 256, 256, 0, stream>>>(x, xh);
    }
    {
        const int groups = NDIM * KDIM / 8;
        cvt_w_f16_kernel<<<groups / 256, 256, 0, stream>>>(wq, zp, wh);
    }

    // WMMA GEMM
    dim3 grid(Mtot / BM, NDIM / BN);              // 32 x 32 blocks
    dim3 block(256);                              // 8 wave32
    ExportWeightBackedLinear_50835232916360_kernel<<<grid, block, 0, stream>>>(
        xh, wh, scale, bias, gamma, out);
}
